// QuantizedLinear_44375602103130
// MI455X (gfx1250) — matrix operand order repaired and checked by reading, not compiled, not run on hardware
//
#include <hip/hip_runtime.h>

// ---------------------------------------------------------------------------
// QuantizedLinear: out[b,s,o] = scale[o] * sum_i x[b,s,i] * w_int[o,i] + bias[o]
// M = B*S = 8192, N = D_OUT = 16384, K = D_IN = 4096
//
// MI455X (gfx1250, wave32) strategy:
//  1) conv_w_kernel: int32 weights -> bf16 in d_ws (exact: |w| <= 127 < 2^8).
//  2) gemm_kernel:   v_wmma_f32_16x16x32_bf16, f32 accumulate, per-channel
//     scale + bias epilogue. Block tile 128x128, K-step 32.
//     - B tile (bf16 weights): Tensor Data Mover DMA (tensor_load_to_lds,
//       TENSORcnt, double-buffered; TDM padding reproduces the 80B pitch).
//       Fallbacks: per-lane global_load_async_to_lds_b128, then reg+ds_store.
//     - A tile: global f32 -> reg -> cvt bf16 -> LDS (conversion needs VALU).
//     - LDS pitch 80B -> conflict-free ds_load_b128 fragment reads.
// ---------------------------------------------------------------------------

#define M_TOT 8192
#define N_TOT 16384
#define K_TOT 4096
#define BM 128
#define BN 128
#define BK 32
#define NKB (K_TOT / BK)        // 128 K-steps
#define LDS_PITCH 80            // 32 bf16 = 64B data + 16B pad per row
#define LDS_A_BYTES (BM * LDS_PITCH)
#define LDS_B_BYTES (BN * LDS_PITCH)

#ifndef __has_builtin
#define __has_builtin(x) 0
#endif

#if __has_builtin(__builtin_amdgcn_tensor_load_to_lds) && \
    __has_builtin(__builtin_amdgcn_s_wait_tensorcnt)
#define USE_TDM 1
#else
#define USE_TDM 0
#endif

#if !USE_TDM && __has_builtin(__builtin_amdgcn_global_load_async_to_lds_b128)
#define USE_ASYNC_LDS 1
#else
#define USE_ASYNC_LDS 0
#endif

#if __has_builtin(__builtin_amdgcn_s_wait_asynccnt)
#define WAIT_ASYNCCNT(n) __builtin_amdgcn_s_wait_asynccnt((unsigned short)(n))
#else
#define WAIT_ASYNCCNT(n) asm volatile("s_wait_asynccnt %0" ::"i"(n) : "memory")
#endif

typedef __bf16 v16bf __attribute__((ext_vector_type(16)));
typedef __bf16 v8bf  __attribute__((ext_vector_type(8)));
typedef __bf16 v4bf  __attribute__((ext_vector_type(4)));
typedef float  v8f   __attribute__((ext_vector_type(8)));
typedef int    v4i   __attribute__((ext_vector_type(4)));
typedef int    v8i   __attribute__((ext_vector_type(8)));
typedef unsigned int v4u __attribute__((ext_vector_type(4)));

typedef __attribute__((address_space(1))) v4i* gv4i_p;
typedef __attribute__((address_space(3))) v4i* lv4i_p;

// --------------------------- weight dequant pre-pass -----------------------
__global__ __launch_bounds__(256) void conv_w_kernel(
    const int* __restrict__ wi, unsigned short* __restrict__ wb) {
  size_t i = (size_t)blockIdx.x * 256 + threadIdx.x;   // 8 elements per thread
  const int4* p = (const int4*)wi + i * 2;
  int4 q0 = p[0];
  int4 q1 = p[1];
  v8bf o;
  o[0] = (__bf16)(float)q0.x; o[1] = (__bf16)(float)q0.y;
  o[2] = (__bf16)(float)q0.z; o[3] = (__bf16)(float)q0.w;
  o[4] = (__bf16)(float)q1.x; o[5] = (__bf16)(float)q1.y;
  o[6] = (__bf16)(float)q1.z; o[7] = (__bf16)(float)q1.w;
  ((v8bf*)wb)[i] = o;
}

// --------------------------------- GEMM ------------------------------------
__global__ __launch_bounds__(256) void gemm_kernel(
    const float* __restrict__ x,            // (M, K) f32
    const unsigned short* __restrict__ wb,  // (N, K) bf16 bits
    const float* __restrict__ scale,        // (N)
    const float* __restrict__ bias,         // (N)
    float* __restrict__ out) {              // (M, N) f32
#if USE_TDM || USE_ASYNC_LDS
  __shared__ __align__(16) unsigned char lds[LDS_A_BYTES + 2 * LDS_B_BYTES];
#else
  __shared__ __align__(16) unsigned char lds[LDS_A_BYTES + LDS_B_BYTES];
#endif
  unsigned char* ldsA = lds;
  unsigned char* ldsB = lds + LDS_A_BYTES;

  const int tid  = threadIdx.x;
  const int wave = tid >> 5;       // 0..7 -> M sub-tile
  const int lane = tid & 31;
  const int lrow = lane & 15;
  const int hi   = lane >> 4;

  const int m0 = blockIdx.y * BM;
  const int n0 = blockIdx.x * BN;

  v8f zero = {};
  v8f acc[8];
#pragma unroll
  for (int i = 0; i < 8; ++i) acc[i] = zero;

  float4 aReg[4];

  auto load_a_glb = [&](int kb) {
#pragma unroll
    for (int j = 0; j < 4; ++j) {
      int c = tid + 256 * j;
      int r = c >> 3, cc = c & 7;
      aReg[j] = *(const float4*)(x + (size_t)(m0 + r) * K_TOT + kb * BK + cc * 4);
    }
  };
  auto store_a_lds = [&]() {
#pragma unroll
    for (int j = 0; j < 4; ++j) {
      int c = tid + 256 * j;
      int r = c >> 3, cc = c & 7;
      v4bf t;
      t[0] = (__bf16)aReg[j].x; t[1] = (__bf16)aReg[j].y;
      t[2] = (__bf16)aReg[j].z; t[3] = (__bf16)aReg[j].w;
      *(v4bf*)(ldsA + r * LDS_PITCH + cc * 8) = t;       // ds_store_b64
    }
  };

  // ---- fragment loaders (ISA 16-bit WMMA layouts, §7.12.2) ----
  auto load_a_frag = [&]() {
    const int arow = wave * 16 + lrow;
    v8bf a0 = *(const v8bf*)(ldsA + arow * LDS_PITCH + hi * 16);
    v8bf a1 = *(const v8bf*)(ldsA + arow * LDS_PITCH + hi * 16 + 32);
    v16bf a;
#pragma unroll
    for (int i = 0; i < 8; ++i) { a[i] = a0[i]; a[i + 8] = a1[i]; }
    return a;
  };
  auto load_b_frag = [&](const unsigned char* bbase, int nt) {
    const int brow = nt * 16 + lrow;
    v8bf b0 = *(const v8bf*)(bbase + brow * LDS_PITCH + hi * 16);
    v8bf b1 = *(const v8bf*)(bbase + brow * LDS_PITCH + hi * 16 + 32);
    v16bf b;
#pragma unroll
    for (int i = 0; i < 8; ++i) { b[i] = b0[i]; b[i + 8] = b1[i]; }
    return b;
  };
  auto compute = [&](const unsigned char* bbase) {
    v16bf a = load_a_frag();
    v16bf bcur = load_b_frag(bbase, 0);
#pragma unroll
    for (int nt = 0; nt < 8; ++nt) {
      v16bf bnext = bcur;
      if (nt < 7) bnext = load_b_frag(bbase, nt + 1);   // 1-deep pipeline
      acc[nt] = __builtin_amdgcn_wmma_f32_16x16x32_bf16(
          false, a, false, bcur, (short)0, acc[nt], false, false);
      bcur = bnext;
    }
  };

#if USE_TDM
  // ---- Tensor Data Mover: one DMA per B tile (128 rows x 32 bf16),
  //      LDS pitch 80B via TDM padding (16 DW data + 4 DW pad per row) ----
  auto issue_tdm_b = [&](int kb, int buf) {
    unsigned lds_off = (unsigned)(size_t)(void*)(ldsB + buf * LDS_B_BYTES);
    unsigned long long ga =
        (unsigned long long)(size_t)(wb + (size_t)n0 * K_TOT + (size_t)kb * BK);
    v4u g0;
    g0[0] = 1u;                                   // count=1 (valid descriptor)
    g0[1] = lds_off;                              // lds_addr
    g0[2] = (unsigned)ga;                         // global_addr[31:0]
    g0[3] = (unsigned)((ga >> 32) & 0x1FFFFFFu)   // global_addr[56:32]
            | (2u << 30);                         // type=2 ("image")
    v8i g1;
    g1[0] = (1 << 16)                             // data_size: 2 bytes
          | (1 << 20)                             // pad_enable
          | (3 << 22)                             // pad_interval: 16 DWORDs
          | (3 << 25);                            // pad_amount:   4 DWORDs
    g1[1] = (int)((K_TOT & 0xFFFF) << 16);        // tensor_dim0 lo (4096)
    g1[2] = (int)(((unsigned)K_TOT >> 16) & 0xFFFFu)
          | (int)((N_TOT & 0xFFFF) << 16);        // dim0 hi | tensor_dim1 lo
    g1[3] = (int)(((unsigned)N_TOT >> 16) & 0xFFFFu)
          | (BK << 16);                           // dim1 hi | tile_dim0 = 32
    g1[4] = BN;                                   // tile_dim1 = 128 (dim2 = 0)
    g1[5] = K_TOT;                                // tensor_dim0_stride lo32
    g1[6] = 0;                                    // dim0_stride hi | dim1_stride lo
    g1[7] = 0;
    v4i gz = {0, 0, 0, 0};                        // groups 2/3 unused (2D)
#if defined(__clang_major__) && __clang_major__ >= 23
    v8i g4 = {0, 0, 0, 0, 0, 0, 0, 0};
    __builtin_amdgcn_tensor_load_to_lds(g0, g1, gz, gz, g4, 0);
#else
    __builtin_amdgcn_tensor_load_to_lds(g0, g1, gz, gz, 0);
#endif
  };

  if (wave == 0) issue_tdm_b(0, 0);
  load_a_glb(0);
  for (int kb = 0; kb < NKB; ++kb) {
    if (wave == 0 && kb + 1 < NKB) issue_tdm_b(kb + 1, (kb + 1) & 1);
    store_a_lds();
    if (wave == 0) {
      if (kb + 1 < NKB) __builtin_amdgcn_s_wait_tensorcnt((unsigned short)1);
      else              __builtin_amdgcn_s_wait_tensorcnt((unsigned short)0);
    }
    __syncthreads();
    if (kb + 1 < NKB) load_a_glb(kb + 1);
    compute(ldsB + (kb & 1) * LDS_B_BYTES);
    __syncthreads();
  }
#elif USE_ASYNC_LDS
  // ---- per-lane async B staging, double buffered ----
  auto issue_async_b = [&](int kb, int buf) {
#pragma unroll
    for (int j = 0; j < 2; ++j) {
      int c = tid + 256 * j;
      int r = c >> 2, cc = c & 3;
      const unsigned short* g =
          wb + (size_t)(n0 + r) * K_TOT + (size_t)kb * BK + cc * 8;
      unsigned char* l = ldsB + buf * LDS_B_BYTES + r * LDS_PITCH + cc * 16;
      __builtin_amdgcn_global_load_async_to_lds_b128(
          (gv4i_p)(void*)g, (lv4i_p)(void*)l, 0, 0);
    }
  };

  issue_async_b(0, 0);
  load_a_glb(0);
  for (int kb = 0; kb < NKB; ++kb) {
    if (kb + 1 < NKB) issue_async_b(kb + 1, (kb + 1) & 1);
    store_a_lds();
    if (kb + 1 < NKB) { WAIT_ASYNCCNT(2); } else { WAIT_ASYNCCNT(0); }
    __syncthreads();
    if (kb + 1 < NKB) load_a_glb(kb + 1);
    compute(ldsB + (kb & 1) * LDS_B_BYTES);
    __syncthreads();
  }
#else
  // ---- fallback: B via global->reg->ds_store, single buffer ----
  uint4 bReg[2];
  auto load_b_glb = [&](int kb) {
#pragma unroll
    for (int j = 0; j < 2; ++j) {
      int c = tid + 256 * j;
      int r = c >> 2, cc = c & 3;
      bReg[j] = *(const uint4*)(wb + (size_t)(n0 + r) * K_TOT + kb * BK + cc * 8);
    }
  };

  load_a_glb(0);
  load_b_glb(0);
  for (int kb = 0; kb < NKB; ++kb) {
    store_a_lds();
#pragma unroll
    for (int j = 0; j < 2; ++j) {
      int c = tid + 256 * j;
      int r = c >> 2, cc = c & 3;
      *(uint4*)(ldsB + r * LDS_PITCH + cc * 16) = bReg[j];
    }
    __syncthreads();
    if (kb + 1 < NKB) { load_a_glb(kb + 1); load_b_glb(kb + 1); }
    compute(ldsB);
    __syncthreads();
  }
#endif

  // ---- epilogue: out = acc * scale[n] + bias[n] ----
#pragma unroll
  for (int nt = 0; nt < 8; ++nt) {
    const int ng = n0 + nt * 16 + lrow;
    const float s  = scale[ng];
    const float bz = bias[ng];
#pragma unroll
    for (int r = 0; r < 8; ++r) {
      const int mg = m0 + wave * 16 + hi * 8 + r;
      out[(size_t)mg * N_TOT + ng] = acc[nt][r] * s + bz;
    }
  }
}

// ------------------------------- launcher ----------------------------------
extern "C" void kernel_launch(void* const* d_in, const int* in_sizes, int n_in,
                              void* d_out, int out_size, void* d_ws, size_t ws_size,
                              hipStream_t stream) {
  const float* x     = (const float*)d_in[0];  // (4, 2048, 4096) f32
  const int*   wi    = (const int*)d_in[1];    // (16384, 4096) int32
  const float* scale = (const float*)d_in[2];  // (16384, 1) f32
  const float* bias  = (const float*)d_in[3];  // (16384,) f32
  float* out = (float*)d_out;                  // (4, 2048, 16384) f32

  unsigned short* wbf = (unsigned short*)d_ws; // 16384*4096 bf16 = 128 MiB

  conv_w_kernel<<<32768, 256, 0, stream>>>(wi, wbf);

  dim3 grid(N_TOT / BN, M_TOT / BM);           // (128, 64)
  gemm_kernel<<<grid, 256, 0, stream>>>(x, wbf, scale, bias, out);

  (void)in_sizes; (void)n_in; (void)out_size; (void)ws_size;
}
